// HFAN_17635135717381
// MI455X (gfx1250) — matrix operand order repaired and checked by reading, not compiled, not run on hardware
//
#include <hip/hip_runtime.h>
#include <hip/hip_bf16.h>
#include <math.h>

typedef __attribute__((ext_vector_type(16))) _Float16 v16h;
typedef __attribute__((ext_vector_type(8)))  _Float16 v8h;
typedef __attribute__((ext_vector_type(8)))  float    v8f;

#define MINF 1e-15f
#define EPSF 1e-6f

// ---------------- WMMA lane-layout helpers (CDNA5 wave32, 16x16x32 f16) ----
// A: 16x32 (MxK) row-major source, lda in halfs (must be multiple of 8, base
// 16B aligned). Per-lane gather = two contiguous 16B vector loads.
__device__ __forceinline__ v16h load_a16(const _Float16* s, int lda) {
  int lane = threadIdx.x & 31;
  int hf = lane >> 4, m = lane & 15;
  const _Float16* base = s + m * lda + hf * 8;
  v8h lo = *(const v8h*)(base);        // k = hf*8 .. hf*8+7
  v8h hi = *(const v8h*)(base + 16);   // k = 16+hf*8 .. 16+hf*8+7
  v16h a;
#pragma unroll
  for (int i = 0; i < 8; i++) { a[i] = lo[i]; a[i + 8] = hi[i]; }
  return a;
}
// B: pre-swizzled layout [tile][lane][16 halfs] -> one 32B contiguous block
// per lane, fully coalesced across the wave.
__device__ __forceinline__ v16h load_b_swz(const _Float16* s, int tile) {
  int lane = threadIdx.x & 31;
  const _Float16* base = s + ((size_t)tile * 32 + lane) * 16;
  v8h lo = *(const v8h*)(base);
  v8h hi = *(const v8h*)(base + 8);
  v16h b;
#pragma unroll
  for (int i = 0; i < 8; i++) { b[i] = lo[i]; b[i + 8] = hi[i]; }
  return b;
}
__device__ __forceinline__ void store_c(float* s, int ldc, const v8f& c) {
  int lane = threadIdx.x & 31;
  int hf = lane >> 4, n = lane & 15;
#pragma unroll
  for (int r = 0; r < 8; r++) s[(r + 8 * hf) * ldc + n] = c[r];
}
__device__ __forceinline__ v8f v8zero() {
  v8f z;
#pragma unroll
  for (int i = 0; i < 8; i++) z[i] = 0.f;
  return z;
}
#define WMMA(a, b, c) \
  __builtin_amdgcn_wmma_f32_16x16x32_f16(false, (a), false, (b), (short)0, (c), false, false)

// ---------------- scalar math helpers -------------------------------------
__device__ __forceinline__ float artanh_(float x) {
  x = fminf(fmaxf(x, -1.f + 1e-7f), 1.f - 1e-7f);
  return 0.5f * (log1pf(x) - log1pf(-x));
}
__device__ __forceinline__ float arcosh_(float x) {
  float t = sqrtf(fmaxf(x * x - 1.f + EPSF, 0.f));
  return logf(x) + log1pf(t / x);
}
__device__ __forceinline__ float sigm_(float x) { return 1.f / (1.f + expf(-x)); }

// ---------------- argument bundles -----------------------------------------
struct IN {
  const float *x, *W1, *b1, *W2, *b2;
  const float *Wihf, *Whhf, *bf, *Wihb, *Whhb, *bb;
  const float *cen, *Wp, *bp, *beta;
};
struct WS {
  _Float16 *P;      // [32000][128] poincare x (f16, K-padded)
  _Float16 *W1B;    // swizzled B: KC=24, NT=7  (86016 halfs)
  _Float16 *WihB;   // swizzled B: 6 x (KC=4, NT=7)  (6*14336)
  _Float16 *WhhB;   // swizzled B: 6 x (KC=4, NT=7)
  _Float16 *WpB;    // swizzled B: KC=7, NT=13 (46592)
  _Float16 *W2B;    // swizzled B: KC=7, NT=48 (172032)
  float *xn;        // [32000] ||x_row||
  float *Vx;        // [6][32000][100] m_matvec(Wih, x)
  float *F;         // [16][20][100][200] GRU outputs
  float *Aatt;      // [32000]
  float *Gam;       // [32000]
  float *Ksc;       // [32000] 2/(1+||f||^2)
  float *cent;      // [201] lorentz_activation(centroid)
};

// swizzled-B index decode: i -> (kc, nt, k-in-chunk, n-in-tile)
__device__ __forceinline__ void swz_pos(size_t i, int NT, int& kc, int& nt,
                                        int& k, int& n) {
  int e = (int)(i & 15);
  int lane = (int)((i >> 4) & 31);
  size_t t = i >> 9;
  nt = (int)(t % NT);
  kc = (int)(t / NT);
  k = (e < 8) ? (((lane >> 4) << 3) + e) : (16 + ((lane >> 4) << 3) + (e - 8));
  n = lane & 15;
}

// ---------------- one_transform row math (Mobius) ---------------------------
// t2 = m_add(m_add(w*mx, vU), bias);  mode 1: sigmoid(logmap0(t2)); mode 0: raw
__device__ void one_transform_row(const float* mxrow, float xn, const float* vU,
                                  const float* bias, float* tmp, float* outr, int mode) {
  float mxn2 = 0.f;
  for (int j = 0; j < 100; j++) mxn2 += mxrow[j] * mxrow[j];
  float mxn = sqrtf(fmaxf(mxn2, MINF));
  float w = tanhf(mxn / xn * artanh_(xn)) / mxn;
  float x2 = 0.f, y2 = 0.f, xy = 0.f;
  for (int j = 0; j < 100; j++) {
    float a = w * mxrow[j], b = vU[j];
    x2 += a * a; y2 += b * b; xy += a * b;
  }
  float den = fmaxf(1.f + 2.f * xy + x2 * y2, MINF);
  float ca = (1.f + 2.f * xy + y2) / den, cb = (1.f - x2) / den;
  for (int j = 0; j < 100; j++) tmp[j] = ca * (w * mxrow[j]) + cb * vU[j];
  x2 = 0.f; y2 = 0.f; xy = 0.f;
  for (int j = 0; j < 100; j++) {
    float a = tmp[j], b = bias[j];
    x2 += a * a; y2 += b * b; xy += a * b;
  }
  den = fmaxf(1.f + 2.f * xy + x2 * y2, MINF);
  ca = (1.f + 2.f * xy + y2) / den; cb = (1.f - x2) / den;
  if (mode == 0) {
    for (int j = 0; j < 100; j++) outr[j] = ca * tmp[j] + cb * bias[j];
  } else {
    float tn2 = 0.f;
    for (int j = 0; j < 100; j++) { float t = ca * tmp[j] + cb * bias[j]; tn2 += t * t; }
    float tn = sqrtf(fmaxf(tn2, MINF));
    float sl = artanh_(tn) / tn;
    for (int j = 0; j < 100; j++) {
      float t = ca * tmp[j] + cb * bias[j];
      outr[j] = sigm_(sl * t);
    }
  }
}

// ================= Kw: weight conversion into swizzled B layouts ============
__global__ void kw_prep(IN in, WS w) {
  size_t tid = (size_t)blockIdx.x * blockDim.x + threadIdx.x;
  size_t stride = (size_t)gridDim.x * blockDim.x;
  int kc, nt, k, n;
  // W1B: logical B[768][112], B[d][j] = W1[j*768+d]
  for (size_t i = tid; i < (size_t)24 * 7 * 512; i += stride) {
    swz_pos(i, 7, kc, nt, k, n);
    int d = kc * 32 + k, j = nt * 16 + n;
    w.W1B[i] = (j < 100) ? (_Float16)in.W1[j * 768 + d] : (_Float16)0.f;
  }
  // WihB/WhhB: 6 mats, logical B[128][112], B[d][j] = W[g][j][d]
  for (size_t i = tid; i < (size_t)6 * 14336; i += stride) {
    int m = (int)(i / 14336);
    size_t rem = i % 14336;
    swz_pos(rem, 7, kc, nt, k, n);
    int d = kc * 32 + k, j = nt * 16 + n;
    int dirr = m / 3, g = m % 3;
    const float* Wi = dirr ? in.Wihb : in.Wihf;
    const float* Wh = dirr ? in.Whhb : in.Whhf;
    bool ok = (d < 100) && (j < 100);
    w.WihB[i] = ok ? (_Float16)Wi[g * 10000 + j * 100 + d] : (_Float16)0.f;
    w.WhhB[i] = ok ? (_Float16)Wh[g * 10000 + j * 100 + d] : (_Float16)0.f;
  }
  // WpB: logical B[224][208], B[d][j] = Wp[j*200+d]
  for (size_t i = tid; i < (size_t)7 * 13 * 512; i += stride) {
    swz_pos(i, 13, kc, nt, k, n);
    int d = kc * 32 + k, j = nt * 16 + n;
    w.WpB[i] = (d < 200 && j < 200) ? (_Float16)in.Wp[j * 200 + d] : (_Float16)0.f;
  }
  // W2B: logical B[224][768], B[d][j] = W2[j*200+d]
  for (size_t i = tid; i < (size_t)7 * 48 * 512; i += stride) {
    swz_pos(i, 48, kc, nt, k, n);
    int d = kc * 32 + k, j = nt * 16 + n;
    w.W2B[i] = (d < 200) ? (_Float16)in.W2[j * 200 + d] : (_Float16)0.f;
  }
  if (blockIdx.x == 0 && threadIdx.x == 0) {
    float r2 = 0.f;
    for (int j = 0; j < 200; j++) r2 += in.cen[j] * in.cen[j];
    float r = sqrtf(r2 + 1e-30f);
    float sh = sinhf(r), ch = coshf(r);
    for (int j = 0; j < 200; j++) w.cent[j] = in.cen[j] / r * sh;
    w.cent[200] = ch;
  }
}

// ================= K1: klein map + W1 GEMM + poincare map ===================
__global__ void k1_proj(IN in, WS w) {
  extern __shared__ char smem[];
  _Float16* XA = (_Float16*)smem;                       // [64][768], 16B aligned
  float* hout = (float*)(smem + 64 * 768 * 2);          // [64][112]
  float* scl  = hout + 64 * 112;                        // [64]
  int tid = threadIdx.x;
  size_t rowbase = (size_t)blockIdx.x * 64;
  if (tid < 64) {
    const float* xr = in.x + (rowbase + tid) * 768;
    float s2 = 0.f;
    for (int j = 0; j < 768; j++) s2 += xr[j] * xr[j];
    float r = sqrtf(s2 + 1e-30f);
    scl[tid] = tanhf(r) / r;   // lorentz_to_klein o lorentz_activation
  }
  __syncthreads();
  for (int idx = tid; idx < 64 * 768; idx += 128) {
    int rr = idx / 768, kk = idx % 768;
    XA[idx] = (_Float16)(in.x[(rowbase + rr) * 768 + kk] * scl[rr]);
  }
  __syncthreads();
  int wv = tid >> 5;
  v8f acc[7];
  for (int i = 0; i < 7; i++) acc[i] = v8zero();
  for (int kc = 0; kc < 24; kc++) {
    v16h a = load_a16(XA + wv * 16 * 768 + kc * 32, 768);
    for (int nt = 0; nt < 7; nt++) {
      v16h b = load_b_swz(w.W1B, kc * 7 + nt);
      acc[nt] = WMMA(a, b, acc[nt]);
    }
  }
  for (int nt = 0; nt < 7; nt++) store_c(hout + wv * 16 * 112 + nt * 16, 112, acc[nt]);
  __syncthreads();
  if (tid < 64) {
    size_t row = rowbase + tid;
    float s2 = 0.f;
    for (int j = 0; j < 100; j++) {
      float h = hout[tid * 112 + j] + in.b1[j];
      hout[tid * 112 + j] = h;
      s2 += h * h;
    }
    float inv = rsqrtf(1.f + s2);       // x = h/sqrt(1+||h||^2)
    w.xn[row] = sqrtf(fmaxf(s2 * inv * inv, MINF));
    for (int j = 0; j < 128; j++)
      w.P[row * 128 + j] = (j < 100) ? (_Float16)(hout[tid * 112 + j] * inv) : (_Float16)0.f;
  }
}

// ================= K2: Vx = m_matvec(Wih[dir][gate], x) for all rows ========
__global__ void k2_ux(WS w) {
  __shared__ float mxb[64 * 112];
  int tid = threadIdx.x, wv = tid >> 5;
  size_t rowbase = (size_t)blockIdx.x * 64;
  int mat = blockIdx.y;
  const _Float16* Bg = w.WihB + (size_t)mat * 14336;
  const _Float16* A = w.P + rowbase * 128;
  v8f acc[7];
  for (int i = 0; i < 7; i++) acc[i] = v8zero();
  for (int kc = 0; kc < 4; kc++) {
    v16h a = load_a16(A + (size_t)wv * 16 * 128 + kc * 32, 128);
    for (int nt = 0; nt < 7; nt++) {
      v16h b = load_b_swz(Bg, kc * 7 + nt);
      acc[nt] = WMMA(a, b, acc[nt]);
    }
  }
  for (int nt = 0; nt < 7; nt++) store_c(mxb + wv * 16 * 112 + nt * 16, 112, acc[nt]);
  __syncthreads();
  if (tid < 64) {
    size_t row = rowbase + tid;
    float xn = w.xn[row];
    float m2 = 0.f;
    for (int j = 0; j < 100; j++) m2 += mxb[tid * 112 + j] * mxb[tid * 112 + j];
    float mn = sqrtf(fmaxf(m2, MINF));
    float s = tanhf(mn / xn * artanh_(xn)) / mn;
    float* o = w.Vx + ((size_t)mat * 32000 + row) * 100;
    for (int j = 0; j < 100; j++) o[j] = s * mxb[tid * 112 + j];
  }
}

// ================= K3: sequential Mobius GRU (one WG per batch x dir) =======
__device__ void gemm_h(const _Float16* A, const _Float16* Bg, float* out) {
  int wv = threadIdx.x >> 5;
  int tiles[7]; int nt_ = 0;
  for (int tt = wv; tt < 49; tt += 8) tiles[nt_++] = tt;
  v8f acc[7];
  for (int i = 0; i < nt_; i++) acc[i] = v8zero();
  for (int kc = 0; kc < 4; kc++)
    for (int i = 0; i < nt_; i++) {
      int mt = tiles[i] / 7, nt = tiles[i] % 7;
      v16h a = load_a16(A + mt * 16 * 128 + kc * 32, 128);
      v16h b = load_b_swz(Bg, kc * 7 + nt);
      acc[i] = WMMA(a, b, acc[i]);
    }
  for (int i = 0; i < nt_; i++) {
    int mt = tiles[i] / 7, nt = tiles[i] % 7;
    store_c(out + mt * 16 * 112 + nt * 16, 112, acc[i]);
  }
}

__global__ void k3_gru(IN in, WS w) {
  extern __shared__ char smem[];
  _Float16* hA = (_Float16*)smem;                     // [112][128] f16 A rows
  float* hf = (float*)(smem + 112 * 128 * 2);         // [100][100] f32 h
  float* mx = hf + 100 * 100;                         // [112][112] gemm out
  float* zb = mx + 112 * 112;                         // [100][100] z gate
  float* sc = zb + 100 * 100;                         // [100][100] scratch
  int tid = threadIdx.x;
  int n = blockIdx.x >> 1, dir = blockIdx.x & 1;
  const float* bias = dir ? in.bb : in.bf;
  const _Float16* WB = w.WhhB + (size_t)dir * 3 * 14336;
  for (int i = tid; i < 112 * 128; i += 256) hA[i] = (_Float16)0.f;
  for (int i = tid; i < 100 * 100; i += 256) hf[i] = 0.f;
  __syncthreads();

  for (int t = 0; t < 20; t++) {
    // ---- z gate: mx = h @ Whh[2]^T
    gemm_h(hA, WB + (size_t)2 * 14336, mx);
    __syncthreads();
    if (tid < 100) {
      int b = tid;
      size_t grow = dir ? ((size_t)n * 2000 + 1999 - t * 100 - b)
                        : ((size_t)n * 2000 + t * 100 + b);
      float x2 = 0.f;
      for (int j = 0; j < 100; j++) x2 += hf[b * 100 + j] * hf[b * 100 + j];
      float xn = sqrtf(fmaxf(x2, MINF));
      one_transform_row(mx + b * 112, xn,
                        w.Vx + ((size_t)(dir * 3 + 2) * 32000 + grow) * 100,
                        bias + 200, sc + b * 100, zb + b * 100, 1);
    }
    __syncthreads();
    // ---- r gate and rh
    gemm_h(hA, WB, mx);
    __syncthreads();
    if (tid < 100) {
      int b = tid;
      size_t grow = dir ? ((size_t)n * 2000 + 1999 - t * 100 - b)
                        : ((size_t)n * 2000 + t * 100 + b);
      float x2 = 0.f;
      for (int j = 0; j < 100; j++) x2 += hf[b * 100 + j] * hf[b * 100 + j];
      float xn = sqrtf(fmaxf(x2, MINF));
      one_transform_row(mx + b * 112, xn,
                        w.Vx + ((size_t)(dir * 3 + 0) * 32000 + grow) * 100,
                        bias + 0, sc + b * 100, sc + b * 100, 1);
      // rh = m_pw(r, h)
      float m2 = 0.f;
      for (int j = 0; j < 100; j++) {
        float v = sc[b * 100 + j] * hf[b * 100 + j];
        m2 += v * v;
      }
      float mn = sqrtf(fmaxf(m2, MINF));
      float s = tanhf(mn / xn * artanh_(xn)) / mn;
      for (int j = 0; j < 100; j++)
        hA[b * 128 + j] = (_Float16)(s * sc[b * 100 + j] * hf[b * 100 + j]);
    }
    __syncthreads();
    // ---- htil = one_transform(Whh[1], rh, ...); update h
    gemm_h(hA, WB + (size_t)1 * 14336, mx);
    __syncthreads();
    if (tid < 100) {
      int b = tid;
      size_t grow = dir ? ((size_t)n * 2000 + 1999 - t * 100 - b)
                        : ((size_t)n * 2000 + t * 100 + b);
      float r2 = 0.f;
      for (int j = 0; j < 100; j++) {
        float v = (float)hA[b * 128 + j];
        r2 += v * v;
      }
      float xnr = sqrtf(fmaxf(r2, MINF));
      one_transform_row(mx + b * 112, xnr,
                        w.Vx + ((size_t)(dir * 3 + 1) * 32000 + grow) * 100,
                        bias + 100, sc + b * 100, sc + b * 100, 0);  // htil -> sc
      // d = m_add(-h, htil)
      float x2 = 0.f, y2 = 0.f, xy = 0.f;
      for (int j = 0; j < 100; j++) {
        float a = -hf[b * 100 + j], c = sc[b * 100 + j];
        x2 += a * a; y2 += c * c; xy += a * c;
      }
      float den = fmaxf(1.f + 2.f * xy + x2 * y2, MINF);
      float ca = (1.f + 2.f * xy + y2) / den, cb = (1.f - x2) / den;
      for (int j = 0; j < 100; j++)
        mx[b * 112 + j] = ca * (-hf[b * 100 + j]) + cb * sc[b * 100 + j];
      // p = m_pw(z, d)
      float dn2 = 0.f, pn2 = 0.f;
      for (int j = 0; j < 100; j++) {
        float d0 = mx[b * 112 + j];
        float pz = zb[b * 100 + j] * d0;
        dn2 += d0 * d0; pn2 += pz * pz;
      }
      float dn = sqrtf(fmaxf(dn2, MINF)), pn = sqrtf(fmaxf(pn2, MINF));
      float sf = tanhf(pn / dn * artanh_(dn)) / pn;
      // hn = m_add(h, p)
      float h2 = 0.f, p2 = 0.f, hp = 0.f;
      for (int j = 0; j < 100; j++) {
        float h0 = hf[b * 100 + j];
        float p = sf * zb[b * 100 + j] * mx[b * 112 + j];
        h2 += h0 * h0; p2 += p * p; hp += h0 * p;
      }
      den = fmaxf(1.f + 2.f * hp + h2 * p2, MINF);
      ca = (1.f + 2.f * hp + p2) / den; cb = (1.f - h2) / den;
      float* Fo = w.F + (((size_t)n * 20 + t) * 100 + b) * 200 + dir * 100;
      for (int j = 0; j < 100; j++) {
        float p = sf * zb[b * 100 + j] * mx[b * 112 + j];
        float hn = ca * hf[b * 100 + j] + cb * p;
        hf[b * 100 + j] = hn;
        hA[b * 128 + j] = (_Float16)hn;
        Fo[j] = hn;
      }
    }
    __syncthreads();
  }
}

// ================= K4a: attention scores (Wp GEMM + lorentz distance) =======
__global__ void k4a_att(IN in, WS w) {
  extern __shared__ char smem[];
  _Float16* XA = (_Float16*)smem;                   // [64][224]
  float* outb = (float*)(smem + 64 * 224 * 2);      // [64][208]
  int tid = threadIdx.x, wv = tid >> 5;
  size_t rowbase = (size_t)blockIdx.x * 64;
  for (int idx = tid; idx < 64 * 224; idx += 128) {
    int rr = idx / 224, kk = idx % 224;
    XA[idx] = (kk < 200) ? (_Float16)w.F[(rowbase + rr) * 200 + kk] : (_Float16)0.f;
  }
  __syncthreads();
  v8f acc[13];
  for (int i = 0; i < 13; i++) acc[i] = v8zero();
  for (int kc = 0; kc < 7; kc++) {
    v16h a = load_a16(XA + wv * 16 * 224 + kc * 32, 224);
    for (int nt = 0; nt < 13; nt++) {
      v16h b = load_b_swz(w.WpB, kc * 13 + nt);
      acc[nt] = WMMA(a, b, acc[nt]);
    }
  }
  for (int nt = 0; nt < 13; nt++) store_c(outb + wv * 16 * 208 + nt * 16, 208, acc[nt]);
  __syncthreads();
  if (tid < 64) {
    size_t row = rowbase + tid;
    const float* fr = w.F + row * 200;
    float f2 = 0.f;
    for (int j = 0; j < 200; j++) f2 += fr[j] * fr[j];
    float fn = sqrtf(fmaxf(f2, MINF));
    float* mxr = outb + tid * 208;
    float m2 = 0.f;
    for (int j = 0; j < 200; j++) m2 += mxr[j] * mxr[j];
    float mn = sqrtf(fmaxf(m2, MINF));
    float s = tanhf(mn / fn * artanh_(fn)) / mn;
    // att = m_add(s*mx, bp)
    float x2 = 0.f, y2 = 0.f, xy = 0.f;
    for (int j = 0; j < 200; j++) {
      float a = s * mxr[j], b = in.bp[j];
      x2 += a * a; y2 += b * b; xy += a * b;
    }
    float den = fmaxf(1.f + 2.f * xy + x2 * y2, MINF);
    float ca = (1.f + 2.f * xy + y2) / den, cb = (1.f - x2) / den;
    for (int j = 0; j < 200; j++) mxr[j] = ca * (s * mxr[j]) + cb * in.bp[j];
    float r2 = 0.f;
    for (int j = 0; j < 200; j++) r2 += mxr[j] * mxr[j];
    float dl = 1.f - r2 + EPSF;
    float pm = ((1.f + r2) / dl) * w.cent[200];
    for (int j = 0; j < 200; j++) pm -= (2.f * mxr[j] / dl) * w.cent[j];
    pm = fminf(fmaxf(pm, 1.f + EPSF), 1e16f);
    w.Aatt[row] = -in.beta[0] * arcosh_(pm) - 1.f;
    float ks = 2.f / (1.f + f2);
    float invg = 1.f - ks * ks * f2;
    invg = fminf(fmaxf(invg, EPSF), 1.f - EPSF);
    w.Gam[row] = fminf(fmaxf(rsqrtf(invg), 1.f + EPSF), 1e16f);
    w.Ksc[row] = ks;
  }
}

// ================= K4b: softmax weights, Klein mean, final GEMM =============
__global__ void k4b_final(IN in, WS w, float* out) {
  __shared__ float aw[2000];
  __shared__ float red[256];
  __shared__ float ob[20 * 200];
  __shared__ alignas(16) _Float16 sA[32 * 224];
  int tid = threadIdx.x, n = blockIdx.x;
  for (int i = tid; i < 2000; i += 256) aw[i] = w.Aatt[(size_t)n * 2000 + i];
  for (int i = tid; i < 32 * 224; i += 256) sA[i] = (_Float16)0.f;
  __syncthreads();
  float m = -1e30f;
  for (int i = tid; i < 2000; i += 256) m = fmaxf(m, aw[i]);
  red[tid] = m;
  __syncthreads();
  for (int s = 128; s > 0; s >>= 1) {
    if (tid < s) red[tid] = fmaxf(red[tid], red[tid + s]);
    __syncthreads();
  }
  float amax = red[0];
  __syncthreads();
  if (tid < 100) {
    float cs = 0.f;
    for (int s = 0; s < 20; s++) {
      int i = s * 100 + tid;
      float v = expf(aw[i] - amax) * w.Gam[(size_t)n * 2000 + i];
      aw[i] = v; cs += v;
    }
    float inv = 1.f / cs;
    for (int s = 0; s < 20; s++) {
      int i = s * 100 + tid;
      aw[i] = aw[i] * inv * w.Ksc[(size_t)n * 2000 + i];  // fold klein scale in
    }
  }
  __syncthreads();
  for (int idx = tid; idx < 4000; idx += 256) {
    int s = idx / 200, d = idx % 200;
    const float* Fb = w.F + (((size_t)n * 20 + s) * 100) * 200 + d;
    float acc = 0.f;
    for (int b = 0; b < 100; b++) acc += aw[s * 100 + b] * Fb[(size_t)b * 200];
    ob[idx] = acc;
  }
  __syncthreads();
  if (tid < 20) {
    float o2 = 0.f;
    for (int d = 0; d < 200; d++) o2 += ob[tid * 200 + d] * ob[tid * 200 + d];
    float sc = 1.f / (1.f + sqrtf(1.f + o2));
    for (int d = 0; d < 200; d++) sA[tid * 224 + d] = (_Float16)(ob[tid * 200 + d] * sc);
  }
  __syncthreads();
  // out[n,s,:] = s_out @ W2^T + b2   (M=32 pad, N=768, K=224 pad)
  int wv = tid >> 5;
  int tiles[12]; int nt_ = 0;
  for (int tt = wv; tt < 96; tt += 8) tiles[nt_++] = tt;
  v8f acc2[12];
  for (int i = 0; i < nt_; i++) acc2[i] = v8zero();
  for (int kc = 0; kc < 7; kc++)
    for (int i = 0; i < nt_; i++) {
      int mt = tiles[i] / 48, nt = tiles[i] % 48;
      v16h a = load_a16(sA + mt * 16 * 224 + kc * 32, 224);
      v16h b = load_b_swz(w.W2B, kc * 48 + nt);
      acc2[i] = WMMA(a, b, acc2[i]);
    }
  int lane = tid & 31, hf = lane >> 4, nn = lane & 15;
  for (int i = 0; i < nt_; i++) {
    int mt = tiles[i] / 48, nt = tiles[i] % 48;
#pragma unroll
    for (int r = 0; r < 8; r++) {
      int s = mt * 16 + r + 8 * hf;
      if (s < 20) {
        int col = nt * 16 + nn;
        out[((size_t)n * 20 + s) * 768 + col] = acc2[i][r] + in.b2[col];
      }
    }
  }
}

// ================= host launcher ============================================
extern "C" void kernel_launch(void* const* d_in, const int* in_sizes, int n_in,
                              void* d_out, int out_size, void* d_ws, size_t ws_size,
                              hipStream_t stream) {
  (void)in_sizes; (void)n_in; (void)out_size; (void)ws_size;
  IN in;
  in.x    = (const float*)d_in[0];
  in.W1   = (const float*)d_in[1];
  in.b1   = (const float*)d_in[2];
  in.W2   = (const float*)d_in[3];
  in.b2   = (const float*)d_in[4];
  in.Wihf = (const float*)d_in[5];
  in.Whhf = (const float*)d_in[6];
  in.bf   = (const float*)d_in[7];
  in.Wihb = (const float*)d_in[8];
  in.Whhb = (const float*)d_in[9];
  in.bb   = (const float*)d_in[10];
  in.cen  = (const float*)d_in[11];
  in.Wp   = (const float*)d_in[12];
  in.bp   = (const float*)d_in[13];
  in.beta = (const float*)d_in[14];

  char* base = (char*)d_ws;
  size_t off = 0;
  auto take = [&](size_t bytes) {
    char* p = base + off;
    off = (off + bytes + 255) & ~(size_t)255;
    return p;
  };
  WS w;
  w.P    = (_Float16*)take((size_t)32000 * 128 * 2);
  w.W1B  = (_Float16*)take((size_t)24 * 7 * 512 * 2);
  w.WihB = (_Float16*)take((size_t)6 * 14336 * 2);
  w.WhhB = (_Float16*)take((size_t)6 * 14336 * 2);
  w.WpB  = (_Float16*)take((size_t)7 * 13 * 512 * 2);
  w.W2B  = (_Float16*)take((size_t)7 * 48 * 512 * 2);
  w.xn   = (float*)take((size_t)32000 * 4);
  w.Vx   = (float*)take((size_t)6 * 32000 * 100 * 4);
  w.F    = (float*)take((size_t)32000 * 200 * 4);
  w.Aatt = (float*)take((size_t)32000 * 4);
  w.Gam  = (float*)take((size_t)32000 * 4);
  w.Ksc  = (float*)take((size_t)32000 * 4);
  w.cent = (float*)take((size_t)201 * 4);

  kw_prep<<<256, 256, 0, stream>>>(in, w);
  size_t lds1 = (size_t)64 * 768 * 2 + (64 * 112 + 64) * 4;
  k1_proj<<<500, 128, lds1, stream>>>(in, w);
  k2_ux<<<dim3(500, 6), 128, 0, stream>>>(w);
  size_t lds3 = (size_t)112 * 128 * 2 +
                ((size_t)100 * 100 + 112 * 112 + 100 * 100 + 100 * 100) * 4;
  k3_gru<<<32, 256, lds3, stream>>>(in, w);
  size_t lds4 = (size_t)64 * 224 * 2 + (size_t)64 * 208 * 4;
  k4a_att<<<500, 128, lds4, stream>>>(in, w);
  k4b_final<<<16, 256, 0, stream>>>(in, w, (float*)d_out);
}
